// Agent_Actor_49881750176087
// MI455X (gfx1250) — compile-verified, matrix operand order checked by reading, not run on hardware
//
#include <hip/hip_runtime.h>
#include <hip/hip_fp16.h>

typedef __attribute__((ext_vector_type(16))) _Float16 v16h;
typedef __attribute__((ext_vector_type(8)))  float    v8f;

#define B_SZ   4096
#define D_SZ   512
#define NACT   6
#define NOPP   3
#define NSAMP  80
#define NKSTEP (D_SZ / 32)      // 16 K-steps of 32
#define NFRAG  (NKSTEP * 2 * 32 * 16)   // 16384 f16 = 32 KB, B-fragment major
#define OUT_DIST_OFS  (B_SZ * NACT)                  // 24576
#define OUT_ENT_OFS   (B_SZ * NACT + NOPP*B_SZ*NACT) // 98304

// ---------------------------------------------------------------------------
// Workspace layout:
//   [0,        32 KB)  : WcFrag f16 — Wcat pre-swizzled into WMMA B-fragment
//                        layout: [kstep 16][ntile 2][lane 32][elem 16]
//   [32 KB, +512 KB)   : L f32 [4096][32]  (x @ Wcat, padded)
// ---------------------------------------------------------------------------

// Pre-swizzle Wcat (= [W_opp cols 0..17 | W[:512] cols 18..23 | pad]) directly
// into the per-lane 16-bit B-fragment layout of v_wmma_f32_16x16x32_f16:
//   lane l carries N = (l&15) + 16*ntile, K-half kb = (l>>4)*8
//   elem e: j=e>>1, kk = kb + (j<4 ? 2j : 16+2(j-4)) + (e&1), K = kstep*32+kk
__global__ void prep_wc(const float* __restrict__ Wopp,
                        const float* __restrict__ W,
                        _Float16* __restrict__ Wc) {
    int i = blockIdx.x * blockDim.x + threadIdx.x;
    if (i >= NFRAG) return;
    int e  = i & 15;
    int l  = (i >> 4) & 31;
    int tl = (i >> 9) & 1;
    int t  = i >> 10;
    int n  = tl * 16 + (l & 15);
    int kb = (l >> 4) * 8;
    int j  = e >> 1, lo = e & 1;
    int kk = kb + ((j < 4) ? 2 * j : 16 + 2 * (j - 4)) + lo;
    int K  = t * 32 + kk;
    float v = 0.f;
    if (n < NOPP * NACT) {                         // opponent weights [3,512,6]
        int k = n / NACT, a = n % NACT;
        v = Wopp[((size_t)k * D_SZ + K) * NACT + a];
    } else if (n < NOPP * NACT + NACT) {           // agent weights W[:512]
        v = W[(size_t)K * NACT + (n - NOPP * NACT)];
    }
    Wc[i] = (_Float16)v;
}

__global__ void zero_ent(float* out) {
    if (threadIdx.x == 0) out[OUT_ENT_OFS] = 0.f;
}

// ---------------------------------------------------------------------------
// Skinny GEMM  L = x @ Wcat  via v_wmma_f32_16x16x32_f16.
// 32 blocks x 8 waves; wave owns one 16-row M-tile. B fragments are single
// aligned 32-byte LDS reads (2x ds_load_b128) thanks to the prep swizzle.
// ---------------------------------------------------------------------------
__global__ __launch_bounds__(256) void gemm_wmma(const float* __restrict__ x,
                                                 const _Float16* __restrict__ Wc,
                                                 float* __restrict__ L) {
    __shared__ __align__(32) _Float16 sWc[NFRAG];   // 32 KB of 320 KB WGP LDS

    {   // cooperative 128-bit staging of the fragment table into LDS
        const uint4* src = (const uint4*)Wc;
        uint4*       dst = (uint4*)sWc;
        for (int i = threadIdx.x; i < NFRAG / 8; i += 256) dst[i] = src[i];
    }
    __syncthreads();

    const int wave = threadIdx.x >> 5;
    const int lane = threadIdx.x & 31;
    const int tile = blockIdx.x * 8 + wave;         // 0..255
    const int row0 = tile * 16;
    const int m    = lane & 15;                     // A: lanes carry M
    const int kb   = (lane >> 4) * 8;               // K-half split
    const float* xr = x + (size_t)(row0 + m) * D_SZ;

    v8f c0 = {}; v8f c1 = {};
    for (int t = 0; t < NKSTEP; ++t) {
        const int k0 = t * 32;
        v16h av;
#pragma unroll
        for (int j = 0; j < 8; ++j) {
            // A 16-bit layout: VGPR j<4 -> K = kb+2j, j>=4 -> K = kb+16+2(j-4)
            int kk = kb + ((j < 4) ? (2 * j) : (16 + 2 * (j - 4)));
            av[2 * j]     = (_Float16)xr[k0 + kk];
            av[2 * j + 1] = (_Float16)xr[k0 + kk + 1];
        }
        const v16h b0 = *(const v16h*)(sWc + ((t * 2 + 0) * 32 + lane) * 16);
        const v16h b1 = *(const v16h*)(sWc + ((t * 2 + 1) * 32 + lane) * 16);
        if (t + 1 < NKSTEP) __builtin_prefetch(xr + k0 + 32, 0, 0);
        c0 = __builtin_amdgcn_wmma_f32_16x16x32_f16(false, av, false, b0,
                                                    (short)0, c0, false, false);
        c1 = __builtin_amdgcn_wmma_f32_16x16x32_f16(false, av, false, b1,
                                                    (short)0, c1, false, false);
    }

    // C/D layout: lane<16 -> M=r, N=lane ; lane>=16 -> M=8+r, N=lane-16
    const int n    = lane & 15;
    const int mofs = (lane >> 4) * 8;
#pragma unroll
    for (int r = 0; r < 8; ++r) {
        float* outp = L + (size_t)(row0 + mofs + r) * 32;
        outp[n]      = c0[r];
        outp[16 + n] = c1[r];
    }
}

// ---------------------------------------------------------------------------
// Per-row epilogue: opponent softmax/entropy/dist, inverse-CDF categorical
// sampling, p2-weighted agent softmax accumulation.
// ---------------------------------------------------------------------------
__device__ __forceinline__ float urand(unsigned s) {
    s = s * 747796405u + 2891336453u;
    s ^= s >> 17; s *= 0xed5ad4bbu;
    s ^= s >> 11; s *= 0xac4c1b51u;
    s ^= s >> 15; s *= 0x31848babu;
    s ^= s >> 14;
    return (float)(s >> 8) * (1.0f / 16777216.0f);
}

__global__ __launch_bounds__(256) void epilogue(const float* __restrict__ L,
                                                const float* __restrict__ bopp,
                                                const float* __restrict__ W,
                                                const float* __restrict__ bvec,
                                                float* __restrict__ out) {
    __shared__ float sWt[NOPP * NACT * NACT];  // W rows 512..529 (one-hot rows)
    __shared__ float sB[NACT];
    __shared__ float sEnt[256];

    if (threadIdx.x < NOPP * NACT * NACT)
        sWt[threadIdx.x] = W[(size_t)D_SZ * NACT + threadIdx.x];
    if (threadIdx.x < NACT) sB[threadIdx.x] = bvec[threadIdx.x];
    __syncthreads();

    const int b = blockIdx.x * blockDim.x + threadIdx.x;   // 0..4095
    float ent = 0.f;
    if (b < B_SZ) {
        const float* Lr = L + (size_t)b * 32;

        float dist[NOPP][NACT], cdf[NOPP][NACT];
#pragma unroll
        for (int k = 0; k < NOPP; ++k) {
            float l[NACT], mx = -1e30f;
#pragma unroll
            for (int a = 0; a < NACT; ++a) {
                l[a] = Lr[k * NACT + a] + bopp[k * NACT + a];
                mx = fmaxf(mx, l[a]);
            }
            float s = 0.f;
#pragma unroll
            for (int a = 0; a < NACT; ++a) { float e = __expf(l[a] - mx); dist[k][a] = e; s += e; }
            const float inv = 1.f / s, ls = __logf(s);
            float c = 0.f;
#pragma unroll
            for (int a = 0; a < NACT; ++a) {
                dist[k][a] *= inv;
                c += dist[k][a];
                cdf[k][a] = c;
                ent -= dist[k][a] * (l[a] - mx - ls);
                out[OUT_DIST_OFS + ((size_t)k * B_SZ + b) * NACT + a] = dist[k][a];
            }
        }

        float xw[NACT];
#pragma unroll
        for (int a = 0; a < NACT; ++a) xw[a] = Lr[NOPP * NACT + a] + sB[a];

        float num[NACT] = {0, 0, 0, 0, 0, 0};
        float den = 0.f;
        for (int s = 0; s < NSAMP; ++s) {
            float p1 = 1.f;
            float al[NACT];
#pragma unroll
            for (int a = 0; a < NACT; ++a) al[a] = xw[a];
#pragma unroll
            for (int k = 0; k < NOPP; ++k) {
                float u = urand(((unsigned)(k * NSAMP + s)) * (unsigned)B_SZ + (unsigned)b);
                int act = 0;
#pragma unroll
                for (int a = 0; a < NACT - 1; ++a) act += (u > cdf[k][a]) ? 1 : 0;
                p1 *= dist[k][act];
                const float* wr = &sWt[(k * NACT + act) * NACT];
#pragma unroll
                for (int a = 0; a < NACT; ++a) al[a] += wr[a];
            }
            float mx = al[0];
#pragma unroll
            for (int a = 1; a < NACT; ++a) mx = fmaxf(mx, al[a]);
            float ssum = 0.f, q[NACT];
#pragma unroll
            for (int a = 0; a < NACT; ++a) { q[a] = __expf(al[a] - mx); ssum += q[a]; }
            const float w = p1 / ssum;          // p1 * softmax == w * q
#pragma unroll
            for (int a = 0; a < NACT; ++a) num[a] += w * q[a];
            den += p1;
        }
        const float invd = 1.f / den;           // p2 normalization folded in
#pragma unroll
        for (int a = 0; a < NACT; ++a) out[(size_t)b * NACT + a] = num[a] * invd;
    }

    sEnt[threadIdx.x] = ent;
    __syncthreads();
    for (int s = 128; s > 0; s >>= 1) {
        if (threadIdx.x < s) sEnt[threadIdx.x] += sEnt[threadIdx.x + s];
        __syncthreads();
    }
    if (threadIdx.x == 0)
        atomicAdd(out + OUT_ENT_OFS, sEnt[0] * (1.0f / (3.0f * (float)B_SZ)));
}

// ---------------------------------------------------------------------------
extern "C" void kernel_launch(void* const* d_in, const int* in_sizes, int n_in,
                              void* d_out, int out_size, void* d_ws, size_t ws_size,
                              hipStream_t stream) {
    const float* x    = (const float*)d_in[0];   // [4096,512]
    const float* Wopp = (const float*)d_in[1];   // [3,512,6]
    const float* bopp = (const float*)d_in[2];   // [3,6]
    const float* W    = (const float*)d_in[3];   // [530,6]
    const float* bv   = (const float*)d_in[4];   // [6]
    float* out = (float*)d_out;

    _Float16* Wc = (_Float16*)d_ws;
    float*    L  = (float*)((char*)d_ws + (size_t)NFRAG * sizeof(_Float16));

    hipLaunchKernelGGL(prep_wc, dim3((NFRAG + 255) / 256), dim3(256), 0, stream,
                       Wopp, W, Wc);
    hipLaunchKernelGGL(zero_ent, dim3(1), dim3(32), 0, stream, out);
    hipLaunchKernelGGL(gemm_wmma, dim3(B_SZ / 16 / 8), dim3(256), 0, stream, x, Wc, L);
    hipLaunchKernelGGL(epilogue, dim3(B_SZ / 256), dim3(256), 0, stream,
                       L, bopp, W, bv, out);
}
